// EncoderGRU_50551765074374
// MI455X (gfx1250) — matrix operand order repaired and checked by reading, not compiled, not run on hardware
//
#include <hip/hip_runtime.h>

typedef _Float16 v16h __attribute__((ext_vector_type(16)));
typedef _Float16 v8h  __attribute__((ext_vector_type(8)));
typedef float    v8f  __attribute__((ext_vector_type(8)));

#define T_STEPS 512
#define BATCH   1024
#define HID     64
#define LDSTR   72   // padded LDS row stride (elements); 144 B = multiple of 16 B

// Load a 16-element f16 fragment as two 16-byte LDS reads (ds_load_b128 x2).
__device__ __forceinline__ v16h load_frag16(const _Float16* p0, const _Float16* p1) {
  v16h r;
  *reinterpret_cast<v8h*>(&r)       = *reinterpret_cast<const v8h*>(p0);
  *(reinterpret_cast<v8h*>(&r) + 1) = *reinterpret_cast<const v8h*>(p1);
  return r;
}

__device__ __forceinline__ v8f wmma16(v16h a, v16h b, v8f c) {
  return __builtin_amdgcn_wmma_f32_16x16x32_f16(false, a, false, b, (short)0, c, false, false);
}

__device__ __forceinline__ float fast_sigmoid(float x) {
  // v_exp_f32 + raw v_rcp_f32 (no IEEE div Newton iterations)
  return __builtin_amdgcn_rcpf(1.f + __expf(-x));
}

__device__ __forceinline__ float fast_tanh(float x) {
#if __has_builtin(__builtin_amdgcn_tanhf)
  return __builtin_amdgcn_tanhf(x);      // gfx1250 v_tanh_f32
#else
  return 2.f * __builtin_amdgcn_rcpf(1.f + __expf(-2.f * x)) - 1.f;
#endif
}

__global__ __launch_bounds__(128)
void gru_fused_kernel(const float* __restrict__ X,
                      const float* __restrict__ W_ih,
                      const float* __restrict__ W_hh,
                      const float* __restrict__ b_ih,
                      const float* __restrict__ b_hh,
                      float* __restrict__ out)
{
  // Weights as f16, row-major [n][k] (B-matrix reads are contiguous along k).
  __shared__ _Float16 wx[3 * HID][LDSTR];
  __shared__ _Float16 wh[3 * HID][LDSTR];
  __shared__ _Float16 xb[16][LDSTR];          // X[t] tile, f16
  __shared__ _Float16 hb[2][16][LDSTR];       // h tile, f16, double-buffered by t parity

  const int tid   = threadIdx.x;
  const int wave  = tid >> 5;                 // 0..3
  const int lane  = tid & 31;
  const int ln    = lane & 15;                // N / A-row index within tile
  const int g     = lane >> 4;                // half-wave select
  const int brow0 = blockIdx.x * 16;          // batch tile base

  // ---- one-time: weights f32 -> f16 into LDS (coalesced) ----
  for (int idx = tid; idx < 3 * HID * HID; idx += 128) {
    int n = idx >> 6, k = idx & 63;
    wx[n][k] = (_Float16)W_ih[idx];
    wh[n][k] = (_Float16)W_hh[idx];
  }
  // zero initial hidden buffer (h0 = 0)
  {
    _Float16* hz = &hb[0][0][0];
    for (int idx = tid; idx < 16 * LDSTR; idx += 128) hz[idx] = (_Float16)0.f;
  }
  __syncthreads();

  // ---- persistent weight B-fragments in VGPRs ----
  // B element e(0..15) of lane = K = q*32 + 16*g + e, column N = tilebase + ln.
  const int jcol = wave * 16 + ln;            // hidden unit owned by this lane
  v16h fxr[2], fxz[2], fxn[2], fhr[2], fhz[2], fhn[2];
#pragma unroll
  for (int q = 0; q < 2; ++q) {
    const int off = q * 32 + g * 16;
    fxr[q] = load_frag16(&wx[jcol      ][off], &wx[jcol      ][off + 8]);
    fxz[q] = load_frag16(&wx[ 64 + jcol][off], &wx[ 64 + jcol][off + 8]);
    fxn[q] = load_frag16(&wx[128 + jcol][off], &wx[128 + jcol][off + 8]);
    fhr[q] = load_frag16(&wh[jcol      ][off], &wh[jcol      ][off + 8]);
    fhz[q] = load_frag16(&wh[ 64 + jcol][off], &wh[ 64 + jcol][off + 8]);
    fhn[q] = load_frag16(&wh[128 + jcol][off], &wh[128 + jcol][off + 8]);
  }

  const float bias_r  = b_ih[jcol]       + b_hh[jcol];
  const float bias_z  = b_ih[ 64 + jcol] + b_hh[ 64 + jcol];
  const float bias_xn = b_ih[128 + jcol];
  const float bias_hn = b_hh[128 + jcol];

  // Hidden state resident in C/D layout: hreg[e] = h[brow0 + e + 8g][jcol]
  v8f hreg = {0.f, 0.f, 0.f, 0.f, 0.f, 0.f, 0.f, 0.f};
  const v8f vzero = {0.f, 0.f, 0.f, 0.f, 0.f, 0.f, 0.f, 0.f};

  // X staging: this thread owns 8 contiguous floats of the 16x64 tile.
  const int xm = (tid * 8) >> 6;
  const int xk = (tid * 8) & 63;
  float4 xr0, xr1;
  {
    const float* p = X + ((long)brow0 + xm) * HID + xk;   // t = 0
    xr0 = *(const float4*)p;
    xr1 = *(const float4*)(p + 4);
  }

  for (int t = 0; t < T_STEPS; ++t) {
    // stage X[t] into LDS as f16
    {
      v8h hx;
      hx[0] = (_Float16)xr0.x; hx[1] = (_Float16)xr0.y;
      hx[2] = (_Float16)xr0.z; hx[3] = (_Float16)xr0.w;
      hx[4] = (_Float16)xr1.x; hx[5] = (_Float16)xr1.y;
      hx[6] = (_Float16)xr1.z; hx[7] = (_Float16)xr1.w;
      *reinterpret_cast<v8h*>(&xb[xm][xk]) = hx;
    }
    __syncthreads();

    // prefetch X[t+1] while WMMAs run
    if (t + 1 < T_STEPS) {
      const float* p = X + ((long)(t + 1) * BATCH + brow0 + xm) * HID + xk;
      xr0 = *(const float4*)p;
      xr1 = *(const float4*)(p + 4);
    }

    // A-fragments: lane row = ln; K runs {q*32+16g .. +15}
    const _Float16* xrow = &xb[ln][0];
    const _Float16* hrow = &hb[t & 1][ln][0];
    v16h ax0 = load_frag16(xrow +      16 * g, xrow +  8 + 16 * g);
    v16h ax1 = load_frag16(xrow + 32 + 16 * g, xrow + 40 + 16 * g);
    v16h ah0 = load_frag16(hrow +      16 * g, hrow +  8 + 16 * g);
    v16h ah1 = load_frag16(hrow + 32 + 16 * g, hrow + 40 + 16 * g);

    // 8 independent WMMA chains, max depth 2 (better ILP at 1 wave/SIMD)
    v8f r0 = {bias_r,  bias_r,  bias_r,  bias_r,  bias_r,  bias_r,  bias_r,  bias_r };
    v8f z0 = {bias_z,  bias_z,  bias_z,  bias_z,  bias_z,  bias_z,  bias_z,  bias_z };
    v8f x0 = {bias_xn, bias_xn, bias_xn, bias_xn, bias_xn, bias_xn, bias_xn, bias_xn};
    v8f h0 = {bias_hn, bias_hn, bias_hn, bias_hn, bias_hn, bias_hn, bias_hn, bias_hn};
    v8f r1 = vzero, z1 = vzero, x1 = vzero, h1 = vzero;

    r0 = wmma16(ax0, fxr[0], r0);   r1 = wmma16(ax1, fxr[1], r1);
    z0 = wmma16(ax0, fxz[0], z0);   z1 = wmma16(ax1, fxz[1], z1);
    x0 = wmma16(ax0, fxn[0], x0);   x1 = wmma16(ax1, fxn[1], x1);
    h0 = wmma16(ah0, fhn[0], h0);   h1 = wmma16(ah1, fhn[1], h1);
    r0 = wmma16(ah0, fhr[0], r0);   r1 = wmma16(ah1, fhr[1], r1);
    z0 = wmma16(ah0, fhz[0], z0);   z1 = wmma16(ah1, fhz[1], z1);

    v8f accr  = r0 + r1;
    v8f accz  = z0 + z1;
    v8f accxn = x0 + x1;
    v8f acchn = h0 + h1;

    // gate math + h update + stores
    float* orow = out + ((long)t * BATCH + brow0) * HID + jcol;
#pragma unroll
    for (int e = 0; e < 8; ++e) {
      const int m  = e + 8 * g;                 // C/D layout row
      float r  = fast_sigmoid(accr[e]);
      float z  = fast_sigmoid(accz[e]);
      float nv = fast_tanh(accxn[e] + r * acchn[e]);
      float hn = nv + z * (hreg[e] - nv);       // (1-z)*n + z*h
      hreg[e] = hn;
      orow[(long)m * HID] = hn;                 // output[t, brow0+m, jcol]
      hb[(t + 1) & 1][m][jcol] = (_Float16)hn;  // A-matrix for next step
    }
    __syncthreads();   // hb[(t+1)&1] complete; xb free for restage
  }

  // h_last = output[T-1] tail region
#pragma unroll
  for (int e = 0; e < 8; ++e) {
    const int m = e + 8 * g;
    out[(long)T_STEPS * BATCH * HID + ((long)brow0 + m) * HID + jcol] = hreg[e];
  }
}

extern "C" void kernel_launch(void* const* d_in, const int* in_sizes, int n_in,
                              void* d_out, int out_size, void* d_ws, size_t ws_size,
                              hipStream_t stream) {
  (void)in_sizes; (void)n_in; (void)d_ws; (void)ws_size; (void)out_size;
  const float* X    = (const float*)d_in[0];
  const float* W_ih = (const float*)d_in[1];
  const float* W_hh = (const float*)d_in[2];
  const float* b_ih = (const float*)d_in[3];
  const float* b_hh = (const float*)d_in[4];
  float* out = (float*)d_out;

  dim3 grid(BATCH / 16);   // 64 workgroups, each owns 16 batch rows for all T
  dim3 block(128);         // 4 wave32 waves
  gru_fused_kernel<<<grid, block, 0, stream>>>(X, W_ih, W_hh, b_ih, b_hh, out);
}
